// MNRE_14800457302379
// MI455X (gfx1250) — matrix operand order repaired and checked by reading, not compiled, not run on hardware
//
#include <hip/hip_runtime.h>
#include <hip/hip_bf16.h>
#include <stdint.h>

// ---------------------------------------------------------------------------
// Fused per-expert MLP ensemble for MI455X (gfx1250).
// All contractions via v_wmma_f32_16x16x32_bf16 (f32 accumulate).
// LDS layouts arranged so every WMMA operand is two contiguous ds_load_b128,
// bank-conflict-free (20-dword padded rows). Weight staging double-buffered.
// ---------------------------------------------------------------------------

typedef __bf16   v16bf __attribute__((ext_vector_type(16)));
typedef float    v8f   __attribute__((ext_vector_type(8)));
typedef float    f4v   __attribute__((ext_vector_type(4)));
typedef uint32_t u4v   __attribute__((ext_vector_type(4)));

constexpr int Bc = 4096, Tc = 32, Xc = 256, Ec = 32, Hc = 512, Lc = 2;
constexpr int BM = 64;                  // batch rows per workgroup
constexpr int KCH_IN = (Tc + Xc) / 32;  // 9 K-chunks, input layer
constexpr int KCH_H  = Hc / 32;         // 16 K-chunks, hidden layers
constexpr int PAD = 20;                 // dwords per 16-dword row (bank-conflict-free)

// LDS layout (bytes)
constexpr int AP_BYTES  = KCH_H * BM * PAD * 4;   // 81920 (80 KB) per act buffer
constexpr int WB_BYTES  = Hc * PAD * 4;           // 40960 (40 KB) per weight buffer
constexpr int OFF_AP0   = 0;
constexpr int OFF_AP1   = AP_BYTES;               // 81920
constexpr int OFF_WB0   = 2 * AP_BYTES;           // 163840
constexpr int OFF_WB1   = OFF_WB0 + WB_BYTES;     // 204800
constexpr int OFF_MISC  = OFF_WB1 + WB_BYTES;     // 245760
constexpr int SMEM_BYTES = OFF_MISC + 4096;       // 249856 <= 320 KB

union ABu { u4v q[2]; uint32_t u[8]; v16bf h; };

// Pack two f32 -> packed bf16 (truncate) with one v_perm_b32:
// D = { hi[31:16], lo[31:16] }.  S0 supplies bytes 4-7, S1 bytes 0-3.
__device__ __forceinline__ uint32_t pack2t(float lo, float hi) {
  return __builtin_amdgcn_perm(__float_as_uint(hi), __float_as_uint(lo),
                               0x07060302u);
}
__device__ __forceinline__ float bf2f_lo(uint32_t u) { return __uint_as_float(u << 16); }
__device__ __forceinline__ float bf2f_hi(uint32_t u) { return __uint_as_float(u & 0xffff0000u); }

// Stage one 32x512 f32 weight chunk into LDS, B-fragment-friendly:
//   wbuf[col*PAD + pk] = pack(W[2*pk][col], W[2*pk+1][col]),  pk in [0,16)
// Each task: 4 cols x 4 pk -> 8 coalesced global b128 loads, 16 v_perm,
// 4 ds_store_b128.
__device__ __forceinline__ void stage_wchunk(const float* __restrict__ Wb,
                                             uint32_t* wbuf, int tid) {
  #pragma unroll
  for (int it = 0; it < 2; ++it) {
    int task = tid + it * 256;          // 0..511
    int q    = task >> 7;               // pk quad 0..3 (rows 8q..8q+7)
    int c4   = (task & 127) << 2;       // col 0..508
    f4v r[8];
    #pragma unroll
    for (int j = 0; j < 8; ++j)
      r[j] = *(const f4v*)(Wb + (size_t)(8 * q + j) * Hc + c4);
    #pragma unroll
    for (int i = 0; i < 4; ++i) {
      u4v d;
      d[0] = pack2t(r[0][i], r[1][i]);
      d[1] = pack2t(r[2][i], r[3][i]);
      d[2] = pack2t(r[4][i], r[5][i]);
      d[3] = pack2t(r[6][i], r[7][i]);
      *(u4v*)(wbuf + (c4 + i) * PAD + 4 * q) = d;
    }
  }
}

__device__ __forceinline__ const float* chunk_ptr(const float* baseA,
                                                  const float* baseB,
                                                  int split, int kc) {
  return (kc < split) ? (baseA + (size_t)kc * 32 * Hc)
                      : (baseB + (size_t)(kc - split) * 32 * Hc);
}

// One GEMM layer: [64,K]x[K,512] + bias + PReLU, repacked bf16 to apdst.
// Weight staging double-buffered (wb0/wb1), one barrier per K-chunk.
__device__ __forceinline__ void layer_gemm(
    const float* __restrict__ baseA, const float* __restrict__ baseB,
    int split, int kchunks,
    const uint32_t* __restrict__ apsrc, uint32_t* __restrict__ apdst,
    uint32_t* __restrict__ wb0, uint32_t* __restrict__ wb1,
    const float* __restrict__ bvec, const float* __restrict__ avec,
    int tid)
{
  const int lane = tid & 31;
  const int wv   = tid >> 5;            // wave 0..7
  const int m0   = (wv & 3) * 16;       // 4 M-waves
  const int n0   = (wv >> 2) * 256;     // 2 N-halves
  const int row  = lane & 15;
  const int hl   = lane >> 4;

  v8f acc[16] = {};                     // 16 N-tiles (128 VGPRs f32 C/D)
  uint32_t* wb[2] = { wb0, wb1 };

  __syncthreads();                      // apsrc ready, weight buffers free
  stage_wchunk(chunk_ptr(baseA, baseB, split, 0), wb0, tid);

  for (int kc = 0; kc < kchunks; ++kc) {
    __syncthreads();                    // stage(kc) done; wb[(kc+1)&1] free
    if (kc + 1 < kchunks)
      stage_wchunk(chunk_ptr(baseA, baseB, split, kc + 1), wb[(kc + 1) & 1], tid);
    if (kc + 2 < kchunks)               // global_prefetch_b8 into L2
      __builtin_prefetch(chunk_ptr(baseA, baseB, split, kc + 2) + tid * 64, 0, 0);

    const uint32_t* wcur = wb[kc & 1];

    // A fragment (16x32 bf16): two contiguous ds_load_b128.
    // pair indices: {4h..4h+3} -> u[0..3],  {8+4h..11+4h} -> u[4..7]
    ABu a;
    const uint32_t* ap = apsrc + (uint32_t)(kc * BM + m0 + row) * PAD + 4 * hl;
    a.q[0] = *(const u4v*)(ap);
    a.q[1] = *(const u4v*)(ap + 8);

    #pragma unroll
    for (int nt = 0; nt < 16; ++nt) {
      // B fragment (32x16 bf16): pairs {4h..4h+3} -> u[0..3], {8+4h..11+4h} -> u[4..7].
      ABu b;
      const uint32_t* wp = wcur + (uint32_t)(n0 + nt * 16 + row) * PAD + 4 * hl;
      b.q[0] = *(const u4v*)(wp);
      b.q[1] = *(const u4v*)(wp + 8);
      acc[nt] = __builtin_amdgcn_wmma_f32_16x16x32_bf16(
          false, a.h, false, b.h, (short)0, acc[nt], false, false);
    }
  }
  __syncthreads();

  // Epilogue: bias + per-channel PReLU, repack bf16 into apdst.
  // C/D layout: row = v + 8*half, col = lane%16.
  uint16_t* dst16 = (uint16_t*)apdst;
  #pragma unroll
  for (int nt = 0; nt < 16; ++nt) {
    int n = n0 + nt * 16 + row;
    int p = n >> 1;
    float bb = bvec[n];
    float aa = avec[n];
    #pragma unroll
    for (int v = 0; v < 8; ++v) {
      float val = acc[nt][v] + bb;
      val = (val >= 0.f) ? val : aa * val;
      int m = m0 + v + 8 * hl;
      uint32_t dw = (uint32_t)((p >> 4) * BM + m) * PAD + (p & 15);
      dst16[dw * 2u + (n & 1)] = (uint16_t)(__float_as_uint(val) >> 16);
    }
  }
}

__global__ __launch_bounds__(256)
void mnre_fused(const float* __restrict__ theta, const float* __restrict__ x,
                const int*   __restrict__ masks,
                const float* __restrict__ W1t, const float* __restrict__ W1x,
                const float* __restrict__ b1,  const float* __restrict__ a1,
                const float* __restrict__ Wh,  const float* __restrict__ bh,
                const float* __restrict__ ah,
                const float* __restrict__ Wo,  const float* __restrict__ bo,
                float* __restrict__ out)
{
  extern __shared__ char smem[];
  uint32_t* ap0   = (uint32_t*)(smem + OFF_AP0);
  uint32_t* ap1   = (uint32_t*)(smem + OFF_AP1);
  uint32_t* wb0   = (uint32_t*)(smem + OFF_WB0);
  uint32_t* wb1   = (uint32_t*)(smem + OFF_WB1);
  float*    maskf = (float*)(smem + OFF_MISC);               // 32 floats
  float*    wo    = (float*)(smem + OFF_MISC + 128);         // 512 floats
  float*    red   = (float*)(smem + OFF_MISC + 128 + 2048);  // 256 floats

  const int tid = threadIdx.x;
  const int e   = blockIdx.x;
  const int b0  = blockIdx.y * BM;

  if (tid < Tc) maskf[tid] = (float)masks[e * Tc + tid];
  __syncthreads();

  // Pack input A = [theta*mask | x] (K=288) into ap0 (padded pair layout).
  #pragma unroll
  for (int it = 0; it < (KCH_IN * 16 * BM) / 256; ++it) {    // 36 iters
    int idx = tid + it * 256;
    int m = idx & 63;
    int p = idx >> 6;                                        // K pair 0..143
    float lo, hi;
    if (p < Tc / 2) {
      const float2 t2 = *(const float2*)(theta + (size_t)(b0 + m) * Tc + 2 * p);
      lo = t2.x * maskf[2 * p];
      hi = t2.y * maskf[2 * p + 1];
    } else {
      const float2 x2 = *(const float2*)(x + (size_t)(b0 + m) * Xc + 2 * (p - Tc / 2));
      lo = x2.x;
      hi = x2.y;
    }
    ap0[(uint32_t)((p >> 4) * BM + m) * PAD + (p & 15)] = pack2t(lo, hi);
  }
  // layer_gemm begins with __syncthreads -> ap0 visible before A loads.

  // Input layer: chunk 0 = W1t (32x512), chunks 1..8 = W1x rows.
  layer_gemm(W1t + (size_t)e * Tc * Hc, W1x + (size_t)e * Xc * Hc,
             1, KCH_IN, ap0, ap1, wb0, wb1, b1 + e * Hc, a1 + e * Hc, tid);

  // Hidden layers (ping-pong ap1 <-> ap0).
  const uint32_t* src = ap1;
  uint32_t* dst = ap0;
  for (int i = 0; i < Lc; ++i) {
    const float* wl = Wh + ((size_t)i * Ec + e) * Hc * Hc;
    layer_gemm(wl, wl, KCH_H, KCH_H, src, dst, wb0, wb1,
               bh + (i * Ec + e) * Hc, ah + (i * Ec + e) * Hc, tid);
    const uint32_t* t = src; src = dst; dst = (uint32_t*)t;
  }
  // final activations now in `src`

  __syncthreads();
  wo[tid]       = Wo[e * Hc + tid];
  wo[tid + 256] = Wo[e * Hc + tid + 256];
  __syncthreads();

  // out[b] = h . Wo + bo : 4 partial dots per row, LDS reduce.
  {
    int m = tid & 63, q = tid >> 6;
    float s = 0.f;
    #pragma unroll 4
    for (int p = q * 64; p < q * 64 + 64; ++p) {             // K pairs
      uint32_t u = src[(uint32_t)((p >> 4) * BM + m) * PAD + (p & 15)];
      s = fmaf(bf2f_lo(u), wo[2 * p],     s);
      s = fmaf(bf2f_hi(u), wo[2 * p + 1], s);
    }
    red[q * 64 + m] = s;
  }
  __syncthreads();
  if (tid < BM) {
    float tot = red[tid] + red[64 + tid] + red[128 + tid] + red[192 + tid] + bo[e];
    out[(size_t)(b0 + tid) * Ec + e] = tot;
  }
}

extern "C" void kernel_launch(void* const* d_in, const int* in_sizes, int n_in,
                              void* d_out, int out_size, void* d_ws, size_t ws_size,
                              hipStream_t stream) {
  (void)in_sizes; (void)n_in; (void)out_size; (void)d_ws; (void)ws_size;
  const float* theta = (const float*)d_in[0];
  const float* x     = (const float*)d_in[1];
  const int*   masks = (const int*)  d_in[2];
  const float* W1t   = (const float*)d_in[3];
  const float* W1x   = (const float*)d_in[4];
  const float* b1    = (const float*)d_in[5];
  const float* a1    = (const float*)d_in[6];
  const float* Wh    = (const float*)d_in[7];
  const float* bh    = (const float*)d_in[8];
  const float* ah    = (const float*)d_in[9];
  const float* Wo    = (const float*)d_in[10];
  const float* bo    = (const float*)d_in[11];
  float* out = (float*)d_out;

  dim3 grid(Ec, Bc / BM);   // 32 experts x 64 batch tiles
  dim3 block(256);          // 8 waves (wave32)
  mnre_fused<<<grid, block, SMEM_BYTES, stream>>>(
      theta, x, masks, W1t, W1x, b1, a1, Wh, bh, ah, Wo, bo, out);
}
